// HiERO_20452634264084
// MI455X (gfx1250) — matrix operand order repaired and checked by reading, not compiled, not run on hardware
//
#include <hip/hip_runtime.h>

// ---------------------------------------------------------------------------
// HiERO U-Net for MI455X (gfx1250): bf16 storage, v_wmma_f32_16x16x32_bf16
// GEMMs with fp32 accumulation, async global->LDS weight staging, fp32 LN.
// ---------------------------------------------------------------------------

typedef unsigned short u16;
typedef __attribute__((ext_vector_type(16))) __bf16 v16bf;
typedef __attribute__((ext_vector_type(8)))  float  v8f;

struct alignas(16) U4 { unsigned a, b, c, d; };
struct alignas(16) F4 { float a, b, c, d; };

union Frag { v16bf v; U4 q[2]; unsigned w[8]; };

__device__ __forceinline__ u16 f2bf(float f) {
  unsigned u = __builtin_bit_cast(unsigned, f);
  u += 0x7fffu + ((u >> 16) & 1u);           // round-to-nearest-even
  return (u16)(u >> 16);
}
__device__ __forceinline__ float bf2f(u16 h) {
  return __builtin_bit_cast(float, (unsigned)h << 16);
}
__device__ __forceinline__ unsigned pack_bf2(float lo, float hi) {
  return (unsigned)f2bf(lo) | ((unsigned)f2bf(hi) << 16);
}

// --- CDNA5 async global->LDS copy (ASYNCcnt tracked), 64B per lane ---------
__device__ __forceinline__ void async_cp_64B(unsigned lds_off, const void* gptr) {
  unsigned long long ga = (unsigned long long)(uintptr_t)gptr;
  asm volatile(
      "global_load_async_to_lds_b128 %0, %1, off\n\t"
      "global_load_async_to_lds_b128 %0, %1, off offset:16\n\t"
      "global_load_async_to_lds_b128 %0, %1, off offset:32\n\t"
      "global_load_async_to_lds_b128 %0, %1, off offset:48"
      :: "v"(lds_off), "v"(ga) : "memory");
}
__device__ __forceinline__ void wait_async0() {
#if __has_builtin(__builtin_amdgcn_s_wait_asynccnt)
  __builtin_amdgcn_s_wait_asynccnt(0);
#else
  asm volatile("s_wait_asynccnt 0x0" ::: "memory");
#endif
}

// A fragment (16x32 bf16, ISA 7.12.2 layout): lane<16 holds row m=lane,
// K chunks [0,8) and [16,24); lane>=16 holds K chunks [8,16) and [24,32).
__device__ __forceinline__ void load_afrag(Frag& f, const u16* rowp, int k0, int hi) {
  const int c0 = k0 + (hi ? 8 : 0);
  f.q[0] = *(const U4*)(rowp + c0);
  f.q[1] = *(const U4*)(rowp + c0 + 16);
}
// fp32 source variant (projection): load 2x8 floats, pack to bf16 in-register.
__device__ __forceinline__ void load_afrag(Frag& f, const float* rowp, int k0, int hi) {
  const int c0 = k0 + (hi ? 8 : 0);
  F4 a0 = *(const F4*)(rowp + c0);
  F4 a1 = *(const F4*)(rowp + c0 + 4);
  F4 b0 = *(const F4*)(rowp + c0 + 16);
  F4 b1 = *(const F4*)(rowp + c0 + 20);
  f.w[0] = pack_bf2(a0.a, a0.b); f.w[1] = pack_bf2(a0.c, a0.d);
  f.w[2] = pack_bf2(a1.a, a1.b); f.w[3] = pack_bf2(a1.c, a1.d);
  f.w[4] = pack_bf2(b0.a, b0.b); f.w[5] = pack_bf2(b0.c, b0.d);
  f.w[6] = pack_bf2(b1.a, b1.b); f.w[7] = pack_bf2(b1.c, b1.d);
}

// B fragment (32x16 bf16): lane<16 holds col n=lane, K [0,8),[16,24); lane>=16 K [8,16),[24,32).
// Served from LDS slice staged per block; row stride 40 bf16 (80B: keeps 16B
// alignment for ds_load_b128, 20-bank stride -> conflict-free half-wave).
constexpr int WSTRIDE = 40;
__device__ __forceinline__ void load_bfrag_lds(Frag& f, const u16* lds_base,
                                               int n_local, int hi) {
  const u16* p = lds_base + n_local * WSTRIDE + (hi ? 8 : 0);
  f.q[0] = *(const U4*)p;
  f.q[1] = *(const U4*)(p + 16);
}

__device__ __forceinline__ v8f wmma_bf16(const Frag& a, const Frag& b, v8f c) {
  return __builtin_amdgcn_wmma_f32_16x16x32_bf16(
      /*neg_a=*/false, a.v, /*neg_b=*/false, b.v,
      /*c_mod=*/(short)0, c, /*reuse_a=*/false, /*reuse_b=*/false);
}

// ---------------------------------------------------------------------------
// GEMM: C[M x 256] = act( A[M x K] @ W1^T  (+ An[M x K] @ W2^T)  + bias )
// block = 256 threads = 8 waves; block tile 64 x 256; wave tile 16 x 128.
// Weight slice (32 x 256) staged to LDS via async copies each K-step.
// ---------------------------------------------------------------------------
template <typename AT, bool HAS_NBR, bool LEAKY>
__global__ __launch_bounds__(256) void k_gemm(
    const AT* __restrict__ A, const u16* __restrict__ An,
    const u16* __restrict__ WT1, const u16* __restrict__ WT2,
    const float* __restrict__ bias, u16* __restrict__ Cout, int M, int K) {
  __shared__ u16 lds1[256 * WSTRIDE];
  __shared__ u16 lds2[HAS_NBR ? 256 * WSTRIDE : 8];

  const int lane = threadIdx.x & 31;
  const int wave = threadIdx.x >> 5;
  const int mw = wave & 3, nw = wave >> 2;
  const int row0 = blockIdx.x * 64 + mw * 16;
  const int n_base = nw * 128;
  const int l15 = lane & 15;
  const int hi = (lane >> 4) & 1;
  const long arow = (long)(row0 + l15) * K;

  // each thread stages one n-column (32 K values = 64B) per K-step
  const int n_st = threadIdx.x;
  const unsigned lds1_off = (unsigned)(uintptr_t)(lds1 + n_st * WSTRIDE);
  const unsigned lds2_off = (unsigned)(uintptr_t)(lds2 + (HAS_NBR ? n_st * WSTRIDE : 0));
  const u16* w1row = WT1 + (long)n_st * K;
  const u16* w2row = HAS_NBR ? (WT2 + (long)n_st * K) : nullptr;

  v8f acc[8] = {};
  for (int k0 = 0; k0 < K; k0 += 32) {
    // --- stage weight slice to LDS (ASYNCcnt path) ---
    async_cp_64B(lds1_off, w1row + k0);
    if (HAS_NBR) async_cp_64B(lds2_off, w2row + k0);
    wait_async0();
    __syncthreads();

    // --- compute ---
    Frag a;  load_afrag(a, A + arow, k0, hi);
    Frag an;
    if (HAS_NBR) load_afrag(an, An + arow, k0, hi);
#pragma unroll
    for (int t = 0; t < 8; ++t) {
      const int nl = n_base + t * 16 + l15;
      Frag b;  load_bfrag_lds(b, lds1, nl, hi);
      acc[t] = wmma_bf16(a, b, acc[t]);
      if (HAS_NBR) {
        Frag b2; load_bfrag_lds(b2, lds2, nl, hi);
        acc[t] = wmma_bf16(an, b2, acc[t]);
      }
    }
    __syncthreads();   // protect LDS slice before next staging
  }

  // C layout: VGPR j, lane -> row = row0 + j + 8*hi, col = n_base + t*16 + l15
#pragma unroll
  for (int t = 0; t < 8; ++t) {
    const int n = n_base + t * 16 + l15;
    const float bs = bias[n];
#pragma unroll
    for (int j = 0; j < 8; ++j) {
      const int m = row0 + j + 8 * hi;
      float c = acc[t][j] + bs;
      if (LEAKY) c = c >= 0.f ? c : 0.2f * c;
      Cout[(long)m * 256 + n] = f2bf(c);
    }
  }
}

// ---------------------------------------------------------------------------
// LayerNorm(+leaky)(+residual): one wave (32 lanes x 8 elems) per 256-wide row.
// ---------------------------------------------------------------------------
template <bool HAS_RES, bool F32OUT>
__global__ __launch_bounds__(256) void k_ln(
    const u16* __restrict__ X, const u16* __restrict__ RES,
    const float* __restrict__ g, const float* __restrict__ b,
    void* __restrict__ OUT, int M) {
  const int row = blockIdx.x * 8 + (threadIdx.x >> 5);
  const int lane = threadIdx.x & 31;
  if (row >= M) return;
  const u16* xr = X + (long)row * 256;
  float v[8], s = 0.f, s2 = 0.f;
#pragma unroll
  for (int j = 0; j < 8; ++j) {
    v[j] = bf2f(xr[lane * 8 + j]);
    s += v[j]; s2 += v[j] * v[j];
  }
#pragma unroll
  for (int off = 16; off >= 1; off >>= 1) {
    s  += __shfl_xor(s,  off, 32);
    s2 += __shfl_xor(s2, off, 32);
  }
  const float mu = s * (1.f / 256.f);
  const float var = s2 * (1.f / 256.f) - mu * mu;
  const float rs = rsqrtf(var + 1e-5f);
#pragma unroll
  for (int j = 0; j < 8; ++j) {
    const int h = lane * 8 + j;
    float o = (v[j] - mu) * rs * g[h] + b[h];
    o = o >= 0.f ? o : 0.2f * o;
    if (HAS_RES) o += bf2f(RES[(long)row * 256 + h]);
    if (F32OUT) ((float*)OUT)[(long)row * 256 + h] = o;
    else        ((u16*)OUT)[(long)row * 256 + h] = f2bf(o);
  }
}

// ---------------------------------------------------------------------------
// Elementwise helpers (bf16 in/out, fp32 math)
// ---------------------------------------------------------------------------
__global__ __launch_bounds__(256) void k_tcast(const float* __restrict__ W,
                                               u16* __restrict__ WT,
                                               int K, int N, long total) {
  long idx = (long)blockIdx.x * blockDim.x + threadIdx.x;
  if (idx >= total) return;
  int k = (int)(idx / N), n = (int)(idx % N);
  WT[(long)n * K + k] = f2bf(W[idx]);
}

__global__ __launch_bounds__(256) void k_ds(const u16* __restrict__ X,
                                            u16* __restrict__ Y,
                                            int Tout, long total) {
  long idx = (long)blockIdx.x * blockDim.x + threadIdx.x;
  if (idx >= total) return;
  int h = (int)(idx & 255);
  long orow = idx >> 8;
  long bb = orow / Tout; int t = (int)(orow % Tout);
  long irow = bb * (long)(2 * Tout) + 2 * t;
  Y[idx] = X[irow * 256 + h];
}

__global__ __launch_bounds__(256) void k_nbr(const u16* __restrict__ X,
                                             u16* __restrict__ Y,
                                             int T, long total) {
  long idx = (long)blockIdx.x * blockDim.x + threadIdx.x;
  if (idx >= total) return;
  int h = (int)(idx & 255);
  long row = idx >> 8;
  long bb = row / T; int t = (int)(row % T);
  long base = bb * (long)T * 256 + h;
  float acc = 0.f, cnt = 0.f;
  if (t - 2 >= 0) { acc += bf2f(X[base + (long)(t - 2) * 256]); cnt += 1.f; }
  if (t - 1 >= 0) { acc += bf2f(X[base + (long)(t - 1) * 256]); cnt += 1.f; }
  if (t + 1 <  T) { acc += bf2f(X[base + (long)(t + 1) * 256]); cnt += 1.f; }
  if (t + 2 <  T) { acc += bf2f(X[base + (long)(t + 2) * 256]); cnt += 1.f; }
  Y[idx] = f2bf(acc / fmaxf(cnt, 1.f));
}

__global__ __launch_bounds__(256) void k_ups(const u16* __restrict__ Cc,
                                             const u16* __restrict__ RES,
                                             u16* __restrict__ OUT,
                                             int Tc, long total) {
  long idx = (long)blockIdx.x * blockDim.x + threadIdx.x;
  if (idx >= total) return;
  int h = (int)(idx & 255);
  long orow = idx >> 8;
  int T2 = Tc * 2;
  long bb = orow / T2; int t = (int)(orow % T2);
  int tc = t >> 1;
  long crow = bb * (long)Tc + tc;
  float val;
  if ((t & 1) == 0)       val = bf2f(Cc[crow * 256 + h]);
  else if (tc == Tc - 1)  val = 0.9f * bf2f(Cc[crow * 256 + h]) +
                                0.1f * bf2f(Cc[(crow - 1) * 256 + h]);
  else                    val = 0.5f * (bf2f(Cc[crow * 256 + h]) +
                                        bf2f(Cc[(crow + 1) * 256 + h]));
  OUT[idx] = f2bf(bf2f(RES[idx]) + val);
}

// ---------------------------------------------------------------------------
// Orchestration
// ---------------------------------------------------------------------------
static inline unsigned nblk(long n) { return (unsigned)((n + 255) / 256); }

extern "C" void kernel_launch(void* const* d_in, const int* /*in_sizes*/, int /*n_in*/,
                              void* d_out, int /*out_size*/, void* d_ws, size_t /*ws_size*/,
                              hipStream_t stream) {
  constexpr int B = 16, T0 = 4096, F = 1024, H = 256;
  constexpr long M0 = (long)B * T0;           // 65536 rows

  const float* x     = (const float*)d_in[0];
  const float* Wp    = (const float*)d_in[6];
  const float* bp    = (const float*)d_in[7];
  const float* Wself = (const float*)d_in[8];   // (6,2,H,H)
  const float* Wnbr  = (const float*)d_in[9];   // (6,2,H,H)
  const float* bconv = (const float*)d_in[10];  // (6,2,H)
  const float* lng   = (const float*)d_in[11];  // (6,H)
  const float* lnb   = (const float*)d_in[12];  // (6,H)

  // --- workspace carve-up (256B aligned) ---
  char* wp = (char*)d_ws;
  auto alloc = [&](size_t bytes) -> char* {
    char* p = wp;
    wp += (bytes + 255) & ~(size_t)255;
    return p;
  };
  u16* WpT   = (u16*)alloc((size_t)F * H * 2);          // 1024x256 ^T
  u16* WsT   = (u16*)alloc((size_t)12 * H * H * 2);     // 12 transposed self mats
  u16* WnT   = (u16*)alloc((size_t)12 * H * H * 2);     // 12 transposed nbr mats
  u16* skip0 = (u16*)alloc((size_t)M0       * H * 2);   // T=4096
  u16* skip1 = (u16*)alloc((size_t)(M0 / 2) * H * 2);   // T=2048
  u16* skip2 = (u16*)alloc((size_t)(M0 / 4) * H * 2);   // T=1024
  u16* bufA  = (u16*)alloc((size_t)M0 * H * 2);
  u16* bufB  = (u16*)alloc((size_t)M0 * H * 2);
  u16* t1b   = (u16*)alloc((size_t)M0 * H * 2);
  u16* t2b   = (u16*)alloc((size_t)M0 * H * 2);
  u16* sb    = (u16*)alloc((size_t)M0 * H * 2);

  // --- weight transpose + bf16 cast ---
  k_tcast<<<nblk((long)F * H), 256, 0, stream>>>(Wp, WpT, F, H, (long)F * H);
  for (int i = 0; i < 12; ++i) {
    k_tcast<<<nblk((long)H * H), 256, 0, stream>>>(Wself + (long)i * H * H,
                                                   WsT + (long)i * H * H, H, H, (long)H * H);
    k_tcast<<<nblk((long)H * H), 256, 0, stream>>>(Wnbr + (long)i * H * H,
                                                   WnT + (long)i * H * H, H, H, (long)H * H);
  }

  // --- projection: feat = x @ Wp + bp  -> skip0 (bf16) ---
  k_gemm<float, false, false><<<(unsigned)(M0 / 64), 256, 0, stream>>>(
      x, nullptr, WpT, nullptr, bp, skip0, (int)M0, F);

  // --- down path ---
  u16* cur = skip0;
  int Tcur = T0;
  for (int d = 0; d < 3; ++d) {
    const int Tn = Tcur / 2;
    const int Mn = B * Tn;
    const long tot = (long)Mn * H;
    const int i0 = d * 2, i1 = d * 2 + 1;

    k_ds<<<nblk(tot), 256, 0, stream>>>(cur, bufA, Tn, tot);                   // f_ds
    k_nbr<<<nblk(tot), 256, 0, stream>>>(bufA, sb, Tn, tot);
    k_gemm<u16, true, true><<<(unsigned)(Mn / 64), 256, 0, stream>>>(
        bufA, sb, WsT + (long)i0 * H * H, WnT + (long)i0 * H * H,
        bconv + i0 * H, t1b, Mn, H);
    k_nbr<<<nblk(tot), 256, 0, stream>>>(t1b, sb, Tn, tot);
    k_gemm<u16, true, false><<<(unsigned)(Mn / 64), 256, 0, stream>>>(
        t1b, sb, WsT + (long)i1 * H * H, WnT + (long)i1 * H * H,
        bconv + i1 * H, t2b, Mn, H);
    u16* outb = (d == 0) ? skip1 : (d == 1) ? skip2 : bufB;
    k_ln<true, false><<<(unsigned)(Mn / 8), 256, 0, stream>>>(
        t2b, bufA, lng + d * H, lnb + d * H, (void*)outb, Mn);   // f_ds + leaky(LN)
    cur = outb;
    Tcur = Tn;
  }

  // --- up path ---
  for (int i = 0; i < 3; ++i) {
    const int s = 3 + i;
    const int Tn = Tcur * 2;
    const int Mn = B * Tn;
    const long tot = (long)Mn * H;
    const int i0 = s * 2, i1 = s * 2 + 1;
    const u16* res = (i == 0) ? skip2 : (i == 1) ? skip1 : skip0;

    k_ups<<<nblk(tot), 256, 0, stream>>>(cur, res, bufA, Tcur, tot);           // res + up2(f)

    if (i == 2) {  // dep == 0 -> neighbor window active
      k_nbr<<<nblk(tot), 256, 0, stream>>>(bufA, sb, Tn, tot);
      k_gemm<u16, true, true><<<(unsigned)(Mn / 64), 256, 0, stream>>>(
          bufA, sb, WsT + (long)i0 * H * H, WnT + (long)i0 * H * H,
          bconv + i0 * H, t1b, Mn, H);
      k_nbr<<<nblk(tot), 256, 0, stream>>>(t1b, sb, Tn, tot);
      k_gemm<u16, true, false><<<(unsigned)(Mn / 64), 256, 0, stream>>>(
          t1b, sb, WsT + (long)i1 * H * H, WnT + (long)i1 * H * H,
          bconv + i1 * H, t2b, Mn, H);
    } else {
      k_gemm<u16, false, true><<<(unsigned)(Mn / 64), 256, 0, stream>>>(
          bufA, nullptr, WsT + (long)i0 * H * H, nullptr,
          bconv + i0 * H, t1b, Mn, H);
      k_gemm<u16, false, false><<<(unsigned)(Mn / 64), 256, 0, stream>>>(
          t1b, nullptr, WsT + (long)i1 * H * H, nullptr,
          bconv + i1 * H, t2b, Mn, H);
    }

    if (i < 2) {
      k_ln<false, false><<<(unsigned)(Mn / 8), 256, 0, stream>>>(
          t2b, nullptr, lng + s * H, lnb + s * H, (void*)bufB, Mn);
      cur = bufB;
    } else {
      k_ln<false, true><<<(unsigned)(Mn / 8), 256, 0, stream>>>(
          t2b, nullptr, lng + s * H, lnb + s * H, d_out, Mn);
    }
    Tcur = Tn;
  }
}
